// Net_66752381715145
// MI455X (gfx1250) — matrix operand order repaired and checked by reading, not compiled, not run on hardware
//
#include <hip/hip_runtime.h>

// GCN: out = A_norm( relu( A_norm(X) W1 + b1 ) ) (W2 Wl) + (b2 Wl + bl)
// A_norm = (D+I)^{-1/2} (A+I) (D+I)^{-1/2}, aggregate-then-transform (exact),
// head folded into layer 2 (no nonlinearity between conv2 and the linear head).

#define IN_F  16
#define H1    512
#define H2    1024
#define OUT_F 16

typedef __attribute__((ext_vector_type(16))) __bf16 v16bf;
typedef __attribute__((ext_vector_type(8)))  float  v8f;
typedef __attribute__((ext_vector_type(4)))  float  v4f;

// ---------------- degree / normalization ----------------

__global__ void deg_init_kernel(float* __restrict__ deg, int n) {
    int i = blockIdx.x * blockDim.x + threadIdx.x;
    if (i < n) deg[i] = 1.0f;              // self-loop contributes 1
}

__global__ void deg_count_kernel(float* __restrict__ deg,
                                 const int* __restrict__ dst, int e) {
    int i = blockIdx.x * blockDim.x + threadIdx.x;
    if (i < e) atomicAdd(&deg[dst[i]], 1.0f);
}

__global__ void rsqrt_kernel(float* __restrict__ deg, int n) {
    int i = blockIdx.x * blockDim.x + threadIdx.x;
    if (i < n) deg[i] = rsqrtf(deg[i]);    // deg -> dinv, in place
}

// ---------------- normalized aggregation ----------------
// Self-loop pass uses plain stores (doubles as accumulator init),
// edge pass uses f32 global atomics (x1 fits in the 192MB L2).

template <int F, int CH>
__global__ void agg_self_kernel(const float* __restrict__ x,
                                float* __restrict__ agg,
                                const float* __restrict__ dinv, int n) {
    const int per = F / CH;
    int t = blockIdx.x * blockDim.x + threadIdx.x;
    int node = t / per, c = t % per;
    if (node >= n) return;
    float w = dinv[node] * dinv[node];
    int base = node * F + c * CH;
#pragma unroll
    for (int j = 0; j < CH; ++j) agg[base + j] = x[base + j] * w;
}

template <int F, int CH>
__global__ void agg_edge_kernel(const float* __restrict__ x,
                                float* __restrict__ agg,
                                const int* __restrict__ src,
                                const int* __restrict__ dst,
                                const float* __restrict__ dinv, int e) {
    const int per = F / CH;
    int t = blockIdx.x * blockDim.x + threadIdx.x;
    int edge = t / per, c = t % per;
    if (edge >= e) return;
    int s = src[edge], d = dst[edge];
    float norm = dinv[s] * dinv[d];
    int bs = s * F + c * CH;
    int bd = d * F + c * CH;
#pragma unroll
    for (int j = 0; j < CH; ++j)
        atomicAdd(&agg[bd + j], x[bs + j] * norm);
}

// ---------------- small helpers ----------------

__global__ void f32_to_bf16_kernel(const float* __restrict__ in,
                                   __bf16* __restrict__ out, int n) {
    int i = blockIdx.x * blockDim.x + threadIdx.x;
    if (i < n) out[i] = (__bf16)in[i];
}

// Wc[i][j] = sum_k W2[i][k] * Wl[k][j]   (512 x 16, exact f32)
__global__ void wc_kernel(const float* __restrict__ W2,
                          const float* __restrict__ Wl,
                          float* __restrict__ Wc) {
    int t = blockIdx.x * blockDim.x + threadIdx.x;
    if (t >= H1 * OUT_F) return;
    int i = t / OUT_F, j = t % OUT_F;
    float acc = 0.0f;
    for (int k = 0; k < H2; ++k) acc += W2[i * H2 + k] * Wl[k * OUT_F + j];
    Wc[t] = acc;
}

// bc[j] = sum_k b2[k] * Wl[k][j] + bl[j]
__global__ void bc_kernel(const float* __restrict__ b2,
                          const float* __restrict__ Wl,
                          const float* __restrict__ bl,
                          float* __restrict__ bc) {
    int j = threadIdx.x;
    if (j >= OUT_F) return;
    float acc = bl[j];
    for (int k = 0; k < H2; ++k) acc += b2[k] * Wl[k * OUT_F + j];
    bc[j] = acc;
}

// ---------------- WMMA GEMM: C = act(A(f32) * B(bf16) + bias) ----------------
// 256-thread block = 8 waves; each wave computes one 16x16 tile (8 consecutive
// M-tiles per block, one N-tile per block). The block's B panel (KT x 16 bf16)
// is DMA'd contiguously into LDS with GLOBAL_LOAD_ASYNC_TO_LDS_B128, then
// transposed/zero-padded once to [n][KTp] so the per-lane B operand is a single
// contiguous 32B LDS vector read per k-step.
//
// Lane layouts used here (wave32; contiguous K per lane half, not the ISA table order):
//   A 16x32 bf16 : M = lane%16; slots j -> K=k0+(lane/16)*16+j, slots 8+j -> +8.
//   B 32x16 bf16 : N = lane%16; slot i -> K = k0 + (lane/16)*16 + i.
//   C/D 16x16 f32: VGPR r -> M = r + 8*(lane/16), N = lane%16.
// Requires M%16==0, N%16==0, K%8==0 (holds: 20000 / {512,16} / {16,512}).
template <int KT, bool RELU>
__global__ __launch_bounds__(256) void wmma_gemm_kernel(
        const float* __restrict__ A, const __bf16* __restrict__ B,
        const float* __restrict__ bias, float* __restrict__ C,
        int M, int N) {
    constexpr int KTp = (KT + 31) & ~31;      // K padded to WMMA depth
    __shared__ __bf16 ldsRaw[KT * 16];        // B panel, [k][n]
    __shared__ __bf16 ldsT[KTp * 16];         // B panel transposed, [n][k]

    const int t      = threadIdx.x;
    const int wave   = t >> 5;
    const int lane   = t & 31;
    const int half   = lane >> 4;
    const int l16    = lane & 15;
    const int tn     = blockIdx.y;
    const int tm     = blockIdx.x * 8 + wave;
    const int Mtiles = M / 16;

    // --- async DMA of B panel into LDS (rows are 32B = two B128 chunks) ---
    const unsigned ldsRawBase = (unsigned)(unsigned long long)&ldsRaw[0];
    const char* Bbase = (const char*)B;
    const int chunks = KT * 2;                // 16-byte chunks
    for (int c = t; c < chunks; c += 256) {
        int k = c >> 1, h = c & 1;
        unsigned loff = ldsRawBase + (unsigned)(c * 16);
        const char* g = Bbase + ((size_t)k * N + (size_t)tn * 16) * 2 + h * 16;
        asm volatile("global_load_async_to_lds_b128 %0, %1, off"
                     :: "v"(loff), "v"(g) : "memory");
    }
    asm volatile("s_wait_asynccnt 0x0" ::: "memory");
    __syncthreads();

    // --- transpose + zero-pad into [n][KTp] ---
    for (int i = t; i < KTp * 16; i += 256) {
        int n = i / KTp, k = i % KTp;
        ldsT[i] = (k < KT) ? ldsRaw[k * 16 + n] : (__bf16)0.0f;
    }
    __syncthreads();

    if (tm < Mtiles) {
        const int    mrow = tm * 16 + l16;
        const int    ncol = tn * 16 + l16;
        const float* Arow = A + (size_t)mrow * KT;

        v8f acc = {};
        for (int k0 = 0; k0 < KTp; k0 += 32) {
            const int ka0 = k0 + half * 16;       // slots 0..7
            const int ka1 = k0 + half * 16 + 8;   // slots 8..15
            v4f a00 = {}, a01 = {}, a10 = {}, a11 = {};
            if (ka0 < KT) {
                a00 = *reinterpret_cast<const v4f*>(Arow + ka0);
                a01 = *reinterpret_cast<const v4f*>(Arow + ka0 + 4);
            }
            if (ka1 < KT) {
                a10 = *reinterpret_cast<const v4f*>(Arow + ka1);
                a11 = *reinterpret_cast<const v4f*>(Arow + ka1 + 4);
            }
            v16bf a;
#pragma unroll
            for (int j = 0; j < 4; ++j) {
                a[j]      = (__bf16)a00[j];
                a[4 + j]  = (__bf16)a01[j];
                a[8 + j]  = (__bf16)a10[j];
                a[12 + j] = (__bf16)a11[j];
            }
            v16bf b = *reinterpret_cast<const v16bf*>(
                &ldsT[l16 * KTp + k0 + half * 16]);

            acc = __builtin_amdgcn_wmma_f32_16x16x32_bf16(
                /*neg_a=*/false, a, /*neg_b=*/false, b,
                /*c_mod=*/(short)0, acc, /*reuse_a=*/false, /*reuse_b=*/false);
        }

        float bn = bias[ncol];
#pragma unroll
        for (int r = 0; r < 8; ++r) {
            int m = tm * 16 + r + 8 * half;
            float v = acc[r] + bn;
            if (RELU) v = fmaxf(v, 0.0f);
            C[(size_t)m * N + ncol] = v;
        }
    }
}

// ---------------- launch ----------------

extern "C" void kernel_launch(void* const* d_in, const int* in_sizes, int n_in,
                              void* d_out, int out_size, void* d_ws, size_t ws_size,
                              hipStream_t stream) {
    const float* x  = (const float*)d_in[0];
    const int*   g  = (const int*)d_in[1];   // graph indices
    const float* W1 = (const float*)d_in[2];
    const float* b1 = (const float*)d_in[3];
    const float* W2 = (const float*)d_in[4];
    const float* b2 = (const float*)d_in[5];
    const float* Wl = (const float*)d_in[6];
    const float* bl = (const float*)d_in[7];
    float* out = (float*)d_out;

    const int n = in_sizes[0] / IN_F;   // 20000 (divisible by 16)
    const int e = in_sizes[1] / 2;      // 160000
    const int* src = g;
    const int* dst = g + e;

    // workspace carve-out (256B aligned), ~84 MB total
    char* ws = (char*)d_ws;
    size_t off = 0;
    auto take = [&](size_t bytes) -> char* {
        char* p = ws + off;
        off = (off + bytes + 255) & ~(size_t)255;
        return p;
    };
    float*  dinv = (float*)take((size_t)n * sizeof(float));
    float*  agg1 = (float*)take((size_t)n * IN_F * sizeof(float));
    float*  x1   = (float*)take((size_t)n * H1 * sizeof(float));
    float*  agg2 = (float*)take((size_t)n * H1 * sizeof(float));
    __bf16* W1bf = (__bf16*)take((size_t)IN_F * H1 * sizeof(__bf16));
    float*  Wc   = (float*)take((size_t)H1 * OUT_F * sizeof(float));
    __bf16* Wcbf = (__bf16*)take((size_t)H1 * OUT_F * sizeof(__bf16));
    float*  bc   = (float*)take(OUT_F * sizeof(float));
    (void)ws_size; (void)n_in; (void)out_size;

    const int B = 256;
    const int Mtiles = n / 16;                 // 1250
    const int mblk   = (Mtiles + 7) / 8;       // 8 M-tiles per block

    // normalization: deg = 1 + in-degree; dinv = rsqrt(deg)
    deg_init_kernel<<<(n + B - 1) / B, B, 0, stream>>>(dinv, n);
    deg_count_kernel<<<(e + B - 1) / B, B, 0, stream>>>(dinv, dst, e);
    rsqrt_kernel<<<(n + B - 1) / B, B, 0, stream>>>(dinv, n);

    // layer 1: agg1 = A_norm X  (16-wide), then x1 = relu(agg1 W1 + b1)
    agg_self_kernel<IN_F, 16><<<(n + B - 1) / B, B, 0, stream>>>(x, agg1, dinv, n);
    agg_edge_kernel<IN_F, 16><<<(e + B - 1) / B, B, 0, stream>>>(x, agg1, src, dst, dinv, e);
    f32_to_bf16_kernel<<<(IN_F * H1 + B - 1) / B, B, 0, stream>>>(W1, W1bf, IN_F * H1);
    {
        dim3 grid(mblk, H1 / 16);
        wmma_gemm_kernel<IN_F, true><<<grid, 256, 0, stream>>>(agg1, W1bf, b1, x1, n, H1);
    }

    // layer 2 aggregation: agg2 = A_norm x1  (512-wide)
    {
        int t_self = n * (H1 / 8);
        int t_edge = e * (H1 / 8);
        agg_self_kernel<H1, 8><<<(t_self + B - 1) / B, B, 0, stream>>>(x1, agg2, dinv, n);
        agg_edge_kernel<H1, 8><<<(t_edge + B - 1) / B, B, 0, stream>>>(x1, agg2, src, dst, dinv, e);
    }

    // fold head into layer 2: Wc = W2 Wl, bc = b2 Wl + bl (exact, no relu between)
    wc_kernel<<<(H1 * OUT_F + B - 1) / B, B, 0, stream>>>(W2, Wl, Wc);
    bc_kernel<<<1, 32, 0, stream>>>(b2, Wl, bl, bc);
    f32_to_bf16_kernel<<<(H1 * OUT_F + B - 1) / B, B, 0, stream>>>(Wc, Wcbf, H1 * OUT_F);

    // out = agg2 Wc + bc
    {
        dim3 grid(mblk, OUT_F / 16);
        wmma_gemm_kernel<H1, false><<<grid, 256, 0, stream>>>(agg2, Wcbf, bc, out, n, OUT_F);
    }
}
